// TensoRF_26508538151542
// MI455X (gfx1250) — matrix operand order repaired and checked by reading, not compiled, not run on hardware
//
#include <hip/hip_runtime.h>
#include <hip/hip_bf16.h>

typedef __attribute__((ext_vector_type(16))) _Float16 v16h;
typedef __attribute__((ext_vector_type(8)))  float    v8f;

#define NRAYS 16384
#define S     128
#define G     128
#define G2    (G * G)
#define G3    (G * G * G)
#define WAVES 8      // 256 threads / wave32

// ---------------------------------------------------------------------------
// A fragment: 16x32 f16 window starting at column kw of an LDS tile whose rows
// are 64 halves. Contiguous K layout used here (not the ISA table order):
//   lanes 0-15 / 16-31 both carry M = lane&15,
//   VGPR v holds K = kw + (lane>=16?16:0) + 2*v .. +1
// ---------------------------------------------------------------------------
__device__ __forceinline__ v16h load_a16(const _Float16* tile, int lane, int kw) {
  const int m    = lane & 15;
  const int hsel = lane >> 4;
  const _Float16* r = tile + m * 64;
  v16h a;
#pragma unroll
  for (int v = 0; v < 8; ++v) {
    const int k = kw + (hsel ? 16 : 0) + 2 * v;
    a[2 * v]     = r[k];
    a[2 * v + 1] = r[k + 1];
  }
  return a;
}

// ---------------------------------------------------------------------------
// B fragment: 32x16 f16 (K window [kw,kw+32), N columns [n0,n0+16)) from an
// LDS weight array laid out [64][ldk] as W[n][k]. Contiguous K layout (same as load_a16):
//   lane n = n0 + (lane&15); lanes 0-15 K=kw+0..15, lanes 16-31 K=kw+16..31.
// ---------------------------------------------------------------------------
__device__ __forceinline__ v16h load_b16(const _Float16* w, int lane, int n0,
                                         int ldk, int kw) {
  const int n  = n0 + (lane & 15);
  const int ks = kw + ((lane >> 4) << 4);
  const _Float16* r = w + n * ldk + ks;
  v16h b;
#pragma unroll
  for (int v = 0; v < 8; ++v) {
    b[2 * v]     = r[2 * v];
    b[2 * v + 1] = r[2 * v + 1];
  }
  return b;
}

__device__ __forceinline__ int iclamp(int v, int lo, int hi) {
  return v < lo ? lo : (v > hi ? hi : v);
}

// ===========================================================================
// Fully fused NeRF kernel: one block (8 wave32) per ray.
//   wave w handles samples [w*16, w*16+16) as one WMMA M=16 tile.
// ===========================================================================
__global__ __launch_bounds__(256)
void nerf_fused(const float* __restrict__ rays, const float* __restrict__ fg,
                const float* __restrict__ Wd1,  const float* __restrict__ bd1,
                const float* __restrict__ Wd2,  const float* __restrict__ bd2,
                const float* __restrict__ Wc1,  const float* __restrict__ bc1,
                const float* __restrict__ Wc2,  const float* __restrict__ bc2,
                float* __restrict__ out) {
  __shared__ _Float16 s_w1d[64 * 32];          // density L1 weights [n][k]
  __shared__ _Float16 s_w1c[64 * 64];          // color   L1 weights [n][k], K padded 35->64
  __shared__ _Float16 s_cin[WAVES][16 * 64];   // per-wave input tile (feats | dirs | 0-pad)
  __shared__ _Float16 s_hid[WAVES][16 * 64];   // per-wave hidden tile (reused dens->color)
  __shared__ float    s_dens[S];
  __shared__ float    s_col[S * 3];

  const int tid  = threadIdx.x;
  const int lane = tid & 31;
  const int wave = tid >> 5;
  const int ray  = blockIdx.x;

  // ---- cooperative weight preload (f32 -> f16, Wc1 zero-padded) ----------
  for (int i = tid; i < 64 * 32; i += 256) s_w1d[i] = (_Float16)Wd1[i];
  for (int i = tid; i < 64 * 64; i += 256) {
    const int n = i >> 6, k = i & 63;
    s_w1c[i] = (_Float16)((k < 35) ? Wc1[n * 35 + k] : 0.0f);
  }
  __syncthreads();

  // ---- per-sample trilinear gather ---------------------------------------
  const float ox = rays[ray * 6 + 0], oy = rays[ray * 6 + 1], oz = rays[ray * 6 + 2];
  const float rdx = rays[ray * 6 + 3], rdy = rays[ray * 6 + 4], rdz = rays[ray * 6 + 5];

  const int m   = lane & 15;              // sample row within the wave tile
  const int ch0 = (lane >> 4) << 4;       // this lane's 16-channel half
  const int sid = wave * 16 + m;          // sample index along the ray
  const float t  = (float)sid * (1.0f / (float)(S - 1));
  const float px = ox + t * rdx, py = oy + t * rdy, pz = oz + t * rdz;

  const float ixf = (px + 1.0f) * 0.5f * (float)(G - 1);
  const float iyf = (py + 1.0f) * 0.5f * (float)(G - 1);
  const float izf = (pz + 1.0f) * 0.5f * (float)(G - 1);
  const float x0f = floorf(ixf), y0f = floorf(iyf), z0f = floorf(izf);
  const float fx = ixf - x0f, fy = iyf - y0f, fz = izf - z0f;
  const int x0 = (int)x0f, y0 = (int)y0f, z0 = (int)z0f;
  const float wx[2] = {1.0f - fx, fx};
  const float wy[2] = {1.0f - fy, fy};
  const float wz[2] = {1.0f - fz, fz};

  float acc[16];
#pragma unroll
  for (int c = 0; c < 16; ++c) acc[c] = 0.0f;

  const float* fgc = fg + (size_t)ch0 * (size_t)G3;
  __builtin_prefetch(fgc, 0, 1);          // gfx1250 global_prefetch_b8

#pragma unroll
  for (int cz = 0; cz < 2; ++cz)
#pragma unroll
    for (int cy = 0; cy < 2; ++cy)
#pragma unroll
      for (int cx = 0; cx < 2; ++cx) {
        const int xi = x0 + cx, yi = y0 + cy, zi = z0 + cz;
        const bool ok = (xi >= 0) & (xi < G) & (yi >= 0) & (yi < G) &
                        (zi >= 0) & (zi < G);
        const float w = wx[cx] * wy[cy] * wz[cz];
        if (ok && w != 0.0f) {
          const int xc = iclamp(xi, 0, G - 1);
          const int yc = iclamp(yi, 0, G - 1);
          const int zc = iclamp(zi, 0, G - 1);
          const size_t base = (size_t)zc * G2 + (size_t)yc * G + (size_t)xc;
#pragma unroll
          for (int c = 0; c < 16; ++c)
            acc[c] += w * fgc[base + (size_t)c * (size_t)G3];
        }
      }

  _Float16* cin = s_cin[wave];
#pragma unroll
  for (int c = 0; c < 16; ++c) cin[m * 64 + ch0 + c] = (_Float16)acc[c];

  // columns 32..63: view dirs then zero padding
  {
    const int cb = 32 + ((lane >> 4) << 4);
#pragma unroll
    for (int j = 0; j < 16; ++j) {
      float v = 0.0f;
      if (lane < 16) {
        if (j == 0) v = rdx; else if (j == 1) v = rdy; else if (j == 2) v = rdz;
      }
      cin[m * 64 + cb + j] = (_Float16)v;
    }
  }

  // ---- density MLP: feats(16x32) @ Wd1^T -> relu -> @ Wd2^T --------------
  const v16h aF = load_a16(s_cin[wave], lane, 0);   // feats, K = 32
  {
#pragma unroll
    for (int nt = 0; nt < 4; ++nt) {
      const v16h b = load_b16(s_w1d, lane, nt * 16, 32, 0);
      v8f h = {};
      h = __builtin_amdgcn_wmma_f32_16x16x32_f16(false, aF, false, b,
                                                 (short)0, h, false, false);
      const int col  = nt * 16 + (lane & 15);
      const float bb = bd1[col];
      const int mb   = (lane >> 4) * 8;
#pragma unroll
      for (int r = 0; r < 8; ++r) {
        const float hv = h[r] + bb;
        s_hid[wave][(mb + r) * 64 + col] = (_Float16)(hv > 0.0f ? hv : 0.0f);
      }
    }
    // layer 2: hidden(16x64) @ Wd2p(64x16), only column n==0 valid
    const v16h h0 = load_a16(s_hid[wave], lane, 0);
    const v16h h1 = load_a16(s_hid[wave], lane, 32);
    const int  n  = lane & 15;
    const int  ks = (lane >> 4) << 4;
    v16h w20, w21;
#pragma unroll
    for (int v = 0; v < 8; ++v) {
      w20[2 * v]     = (_Float16)((n == 0) ? Wd2[ks + 2 * v]          : 0.0f);
      w20[2 * v + 1] = (_Float16)((n == 0) ? Wd2[ks + 2 * v + 1]      : 0.0f);
      w21[2 * v]     = (_Float16)((n == 0) ? Wd2[32 + ks + 2 * v]     : 0.0f);
      w21[2 * v + 1] = (_Float16)((n == 0) ? Wd2[32 + ks + 2 * v + 1] : 0.0f);
    }
    v8f d = {};
    d = __builtin_amdgcn_wmma_f32_16x16x32_f16(false, h0, false, w20,
                                               (short)0, d, false, false);
    d = __builtin_amdgcn_wmma_f32_16x16x32_f16(false, h1, false, w21,
                                               (short)0, d, false, false);
    if (n == 0) {
      const float b2 = bd2[0];
      const int mb = (lane >> 4) * 8;
#pragma unroll
      for (int r = 0; r < 8; ++r) s_dens[wave * 16 + mb + r] = d[r] + b2;
    }
  }

  // ---- color MLP: cin(16x64) @ Wc1p^T -> relu -> @ Wc2p^T ----------------
  {
    const v16h c1 = load_a16(s_cin[wave], lane, 32); // dirs + pad, K 32..63
#pragma unroll
    for (int nt = 0; nt < 4; ++nt) {
      const v16h b0 = load_b16(s_w1c, lane, nt * 16, 64, 0);
      const v16h b1 = load_b16(s_w1c, lane, nt * 16, 64, 32);
      v8f h = {};
      h = __builtin_amdgcn_wmma_f32_16x16x32_f16(false, aF, false, b0,
                                                 (short)0, h, false, false);
      h = __builtin_amdgcn_wmma_f32_16x16x32_f16(false, c1, false, b1,
                                                 (short)0, h, false, false);
      const int col  = nt * 16 + (lane & 15);
      const float bb = bc1[col];
      const int mb   = (lane >> 4) * 8;
#pragma unroll
      for (int r = 0; r < 8; ++r) {
        const float hv = h[r] + bb;
        s_hid[wave][(mb + r) * 64 + col] = (_Float16)(hv > 0.0f ? hv : 0.0f);
      }
    }
    // layer 2: hidden(16x64) @ Wc2p(64x16), columns n in {0,1,2} valid
    const v16h h0 = load_a16(s_hid[wave], lane, 0);
    const v16h h1 = load_a16(s_hid[wave], lane, 32);
    const int  n  = lane & 15;
    const int  ks = (lane >> 4) << 4;
    v16h w20, w21;
#pragma unroll
    for (int v = 0; v < 8; ++v) {
      w20[2 * v]     = (_Float16)((n < 3) ? Wc2[n * 64 + ks + 2 * v]          : 0.0f);
      w20[2 * v + 1] = (_Float16)((n < 3) ? Wc2[n * 64 + ks + 2 * v + 1]      : 0.0f);
      w21[2 * v]     = (_Float16)((n < 3) ? Wc2[n * 64 + 32 + ks + 2 * v]     : 0.0f);
      w21[2 * v + 1] = (_Float16)((n < 3) ? Wc2[n * 64 + 32 + ks + 2 * v + 1] : 0.0f);
    }
    v8f d = {};
    d = __builtin_amdgcn_wmma_f32_16x16x32_f16(false, h0, false, w20,
                                               (short)0, d, false, false);
    d = __builtin_amdgcn_wmma_f32_16x16x32_f16(false, h1, false, w21,
                                               (short)0, d, false, false);
    if (n < 3) {
      const float bb = bc2[n];
      const int mb = (lane >> 4) * 8;
#pragma unroll
      for (int r = 0; r < 8; ++r)
        s_col[(wave * 16 + mb + r) * 3 + n] = d[r] + bb;
    }
  }

  // ---- volume rendering: sequential cumprod scan over the ray ------------
  __syncthreads();
  if (tid == 0) {
    float T = 1.0f, cr = 0.0f, cg = 0.0f, cb = 0.0f;
    const float dt = 1.0f / (float)(S - 1);
#pragma unroll 4
    for (int i = 0; i < S; ++i) {
      const float delta = (i == S - 1) ? 1.0e10f : dt;
      const float alpha = 1.0f - __expf(-s_dens[i] * delta);
      const float w = alpha * T;
      cr += w * s_col[i * 3 + 0];
      cg += w * s_col[i * 3 + 1];
      cb += w * s_col[i * 3 + 2];
      T *= (1.0f - alpha + 1e-10f);
    }
    out[ray * 3 + 0] = cr;
    out[ray * 3 + 1] = cg;
    out[ray * 3 + 2] = cb;
  }
}

extern "C" void kernel_launch(void* const* d_in, const int* in_sizes, int n_in,
                              void* d_out, int out_size, void* d_ws, size_t ws_size,
                              hipStream_t stream) {
  (void)in_sizes; (void)n_in; (void)d_ws; (void)ws_size; (void)out_size;
  nerf_fused<<<dim3(NRAYS), dim3(256), 0, stream>>>(
      (const float*)d_in[0], (const float*)d_in[1],
      (const float*)d_in[2], (const float*)d_in[3],
      (const float*)d_in[4], (const float*)d_in[5],
      (const float*)d_in[6], (const float*)d_in[7],
      (const float*)d_in[8], (const float*)d_in[9],
      (float*)d_out);
}
